// RPNModule_33483565040154
// MI455X (gfx1250) — matrix operand order repaired and checked by reading, not compiled, not run on hardware
//
#include <hip/hip_runtime.h>
#include <hip/hip_bf16.h>
#include <math.h>

typedef __attribute__((ext_vector_type(16))) _Float16 v16h;
typedef __attribute__((ext_vector_type(8)))  _Float16 v8h;
typedef __attribute__((ext_vector_type(8)))  float    v8f;

#define NEG_INF_F   (-1e9f)
#define BBOX_CLIP_F 4.135166556742356f   // log(1000/16)
#define NMS_THR     0.7f

// ---------------- weight prep ----------------
// wB[n][k] with k = r*256 + ci (r = ky*3+kx), f16. conv_w is (O=256, I=256, 3, 3).
__global__ void prep_conv_weights(const float* __restrict__ w, _Float16* __restrict__ wB) {
    int idx = blockIdx.x * blockDim.x + threadIdx.x;   // 256*2304
    if (idx >= 256 * 2304) return;
    int n  = idx / 2304;
    int k  = idx - n * 2304;
    int r  = k >> 8;      // tap 0..8
    int ci = k & 255;
    wB[idx] = (_Float16)w[n * 2304 + ci * 9 + r];
}

// hB[n2][k]: n2 0..2 = cls_w rows, 3..14 = bbox_w rows, 15 = zero. hbias[16].
__global__ void prep_head_weights(const float* __restrict__ cls_w, const float* __restrict__ cls_b,
                                  const float* __restrict__ bbox_w, const float* __restrict__ bbox_b,
                                  _Float16* __restrict__ hB, float* __restrict__ hbias) {
    int idx = blockIdx.x * blockDim.x + threadIdx.x;   // 16*256
    if (idx < 16)
        hbias[idx] = (idx < 3) ? cls_b[idx] : (idx < 15 ? bbox_b[idx - 3] : 0.0f);
    if (idx >= 16 * 256) return;
    int n = idx >> 8, k = idx & 255;
    float v = (n < 3) ? cls_w[n * 256 + k] : (n < 15 ? bbox_w[(n - 3) * 256 + k] : 0.0f);
    hB[idx] = (_Float16)v;
}

// ---------------- padded-halo helpers ----------------
// Zero the 1-pixel halo ring of the padded NHWC f16 buffer [2][H+2][W+2][256].
__global__ void pad_halo_zero(_Float16* __restrict__ out, int H, int W) {
    int Hp = H + 2, Wp = W + 2;
    int totpix = 2 * Hp * Wp;
    for (int pix = blockIdx.x * blockDim.x + threadIdx.x; pix < totpix;
         pix += gridDim.x * blockDim.x) {
        int b  = pix / (Hp * Wp);
        int r2 = pix - b * Hp * Wp;
        int yy = r2 / Wp, xx = r2 - yy * Wp;
        if (yy == 0 || yy == Hp - 1 || xx == 0 || xx == Wp - 1) {
            v8h z = {};
            v8h* dst = (v8h*)(out + (size_t)pix * 256);
#pragma unroll
            for (int c = 0; c < 32; ++c) dst[c] = z;
        }
    }
}

// NCHW fp32 -> padded NHWC f16 interior (reads coalesced over the NCHW input).
__global__ void nchw_to_nhwc_pad_f16(const float* __restrict__ in, _Float16* __restrict__ out,
                                     int H, int W) {
    int HW = H * W;
    int Hp = H + 2, Wp = W + 2;
    int total = 2 * 256 * HW;
    for (int idx = blockIdx.x * blockDim.x + threadIdx.x; idx < total;
         idx += gridDim.x * blockDim.x) {
        int b   = idx / (256 * HW);
        int r0  = idx - b * 256 * HW;
        int c   = r0 / HW;
        int pix = r0 - c * HW;
        int y = pix / W, x = pix - y * W;
        out[(((size_t)(b * Hp + y + 1)) * Wp + (x + 1)) * 256 + c] = (_Float16)in[idx];
    }
}

// ---------------- 3x3 conv (implicit GEMM, WMMA f16) + bias + ReLU ----------------
// xp: padded NHWC f16 [2,H+2,W+2,256]; wB: [256][2304] f16 (B^T); t out: NHWC f16 (unpadded).
// Wave tile: 32 pixels (2 M-frags) x 64 out-channels (4 N-frags) = 8 accumulators.
// Per k-step: 4 A b128-loads + 8 B b128-loads feed 8 WMMAs (branch-free via halo).
__global__ void __launch_bounds__(256)
conv3x3_wmma(const _Float16* __restrict__ xp, const _Float16* __restrict__ wB,
             const float* __restrict__ bias, _Float16* __restrict__ t, int H, int W) {
    const int C = 256;
    const int HWp = H * W;
    const int P = 2 * HWp;
    const int Wp = W + 2, Hp = H + 2;
    const int mt2 = P >> 5;                 // 32-pixel tiles
    int wave = blockIdx.x * 8 + (threadIdx.x >> 5);
    if (wave >= mt2 * 4) return;
    int m2 = wave >> 2;                     // which 32-pixel tile
    int n4 = wave & 3;                      // which 64-channel group
    int lane = threadIdx.x & 31;
    int lm = lane & 15, hs = lane >> 4;

    // A: two pixel rows per lane (M fragment rows = lane%16)
    int p0 = m2 * 32 + lm;
    int p1 = p0 + 16;
    int b0 = p0 / HWp, q0 = p0 - b0 * HWp, y0 = q0 / W, x0 = q0 - y0 * W;
    int b1 = p1 / HWp, q1 = p1 - b1 * HWp, y1 = q1 / W, x1 = q1 - y1 * W;
    const _Float16* a0 = xp + (((size_t)(b0 * Hp + y0)) * Wp + x0) * C + hs * 16;
    const _Float16* a1 = xp + (((size_t)(b1 * Hp + y1)) * Wp + x1) * C + hs * 16;

    // B: four 16-channel fragments, n = n4*64 + f*16 + lm (N = lane%16)
    const _Float16* brow = wB + (size_t)(n4 * 64 + lm) * 2304 + hs * 16;

    v8f acc[8] = {};                        // acc[f] for p0-rows, acc[4+f] for p1-rows
    for (int r = 0; r < 9; ++r) {
        int toff = ((r / 3) * Wp + (r % 3)) * C;   // uniform tap offset (halo => no bounds)
        const _Float16* A0 = a0 + toff;
        const _Float16* A1 = a1 + toff;
        if (r < 8) __builtin_prefetch(brow + (r + 1) * 256, 0, 3);
        for (int c0 = 0; c0 < 256; c0 += 32) {
            v8h lo0 = *(const v8h*)(A0 + c0);
            v8h hi0 = *(const v8h*)(A0 + c0 + 8);
            v8h lo1 = *(const v8h*)(A1 + c0);
            v8h hi1 = *(const v8h*)(A1 + c0 + 8);
            v16h fa0 = __builtin_shufflevector(lo0, hi0, 0,1,2,3,4,5,6,7,8,9,10,11,12,13,14,15);
            v16h fa1 = __builtin_shufflevector(lo1, hi1, 0,1,2,3,4,5,6,7,8,9,10,11,12,13,14,15);
#pragma unroll
            for (int f = 0; f < 4; ++f) {
                v16h bm = *(const v16h*)(brow + (size_t)f * 16 * 2304 + r * 256 + c0);
                acc[f]     = __builtin_amdgcn_wmma_f32_16x16x32_f16(false, fa0, false, bm,
                                                                    (short)0, acc[f], false, false);
                acc[4 + f] = __builtin_amdgcn_wmma_f32_16x16x32_f16(false, fa1, false, bm,
                                                                    (short)0, acc[4 + f], false, false);
            }
        }
    }
#pragma unroll
    for (int f = 0; f < 4; ++f) {
        int n = n4 * 64 + f * 16 + lm;      // D: N = lane%16
        float bv = bias[n];
#pragma unroll
        for (int i = 0; i < 8; ++i) {       // D: M = i + (lane/16)*8
            int M = i + hs * 8;
            float v0 = acc[f][i] + bv;      v0 = v0 > 0.0f ? v0 : 0.0f;
            float v1 = acc[4 + f][i] + bv;  v1 = v1 > 0.0f ? v1 : 0.0f;
            t[(size_t)(m2 * 32 + M) * C + n]      = (_Float16)v0;
            t[(size_t)(m2 * 32 + 16 + M) * C + n] = (_Float16)v1;
        }
    }
}

// ---------------- head GEMM (WMMA) + anchor decode + clip ----------------
__global__ void __launch_bounds__(256)
head_wmma(const _Float16* __restrict__ t, const _Float16* __restrict__ hB,
          const float* __restrict__ hbias, float* __restrict__ scores,
          float* __restrict__ boxes, int H, int W, float stride, float asize) {
    const int C = 256;
    const int HWp = H * W;
    const int P = 2 * HWp;
    const int m_tiles = P >> 4;
    int wid = threadIdx.x >> 5, lane = threadIdx.x & 31;
    int lm = lane & 15, hs = lane >> 4;
    int m_tile = blockIdx.x * 8 + wid;
    bool validw = m_tile < m_tiles;

    __shared__ float hls[8][16][17];

    if (validw) {
        int p = m_tile * 16 + lm;
        const _Float16* abase = t + (size_t)p * C;
        const _Float16* brow = hB + lm * 256 + hs * 16;
        v8f acc = {};
        for (int c0 = 0; c0 < 256; c0 += 32) {
            v8h alo = *(const v8h*)(abase + c0 + hs * 16);
            v8h ahi = *(const v8h*)(abase + c0 + 8 + hs * 16);
            v16h a = __builtin_shufflevector(alo, ahi, 0,1,2,3,4,5,6,7,8,9,10,11,12,13,14,15);
            v16h bm = *(const v16h*)(brow + c0);
            acc = __builtin_amdgcn_wmma_f32_16x16x32_f16(false, a, false, bm,
                                                         (short)0, acc, false, false);
        }
        float bv = hbias[lm];
#pragma unroll
        for (int i = 0; i < 8; ++i)
            hls[wid][i + hs * 8][lm] = acc[i] + bv;
    }
    __syncthreads();

    for (int s = threadIdx.x; s < 128; s += 256) {
        int w = s >> 4, mloc = s & 15;
        int mt = blockIdx.x * 8 + w;
        if (mt >= m_tiles) continue;
        int p = mt * 16 + mloc;
        int b = p / HWp;
        int rem = p - b * HWp;
        int y = rem / W;
        int x = rem - y * W;
        const float* o = hls[w][mloc];
        int nbase = b * (HWp * 3) + rem * 3;
        float cx = x * stride + 0.5f, cy = y * stride + 0.5f;
#pragma unroll
        for (int a = 0; a < 3; ++a) {
            float ratio = (a == 0) ? 0.5f : (a == 1) ? 1.0f : 2.0f;
            float wa = asize * sqrtf(1.0f / ratio);
            float ha = asize * sqrtf(ratio);
            float widths = wa + 1.0f, heights = ha + 1.0f;
            float dx = o[3 + a * 4 + 0];
            float dy = o[3 + a * 4 + 1];
            float dw = fminf(o[3 + a * 4 + 2], BBOX_CLIP_F);
            float dh = fminf(o[3 + a * 4 + 3], BBOX_CLIP_F);
            float pcx = dx * widths + cx, pcy = dy * heights + cy;
            float pw = expf(dw) * widths, ph = expf(dh) * heights;
            float x1 = pcx - 0.5f * pw, y1 = pcy - 0.5f * ph;
            float x2 = pcx + 0.5f * pw - 1.0f, y2 = pcy + 0.5f * ph - 1.0f;
            x1 = fminf(fmaxf(x1, 0.0f), 1023.0f);
            y1 = fminf(fmaxf(y1, 0.0f), 1023.0f);
            x2 = fminf(fmaxf(x2, 0.0f), 1023.0f);
            y2 = fminf(fmaxf(y2, 0.0f), 1023.0f);
            int idx = nbase + a;
            scores[idx] = o[a];
            float* bo = boxes + (size_t)idx * 4;
            bo[0] = x1; bo[1] = y1; bo[2] = x2; bo[3] = y2;
        }
    }
}

// ---------------- per-(batch,level) top-k + greedy NMS ----------------
__device__ __forceinline__ unsigned fkey(float f) {
    unsigned bits = __float_as_uint(f);
    return bits ^ ((bits >> 31) ? 0xFFFFFFFFu : 0x80000000u);   // monotone: larger f -> larger u
}

__global__ void __launch_bounds__(1024)
topk_nms(const float* __restrict__ scores, const float* __restrict__ boxes,
         int N, int k, float* __restrict__ stg_s, float* __restrict__ stg_b,
         int off, int TOT) {
    int b = blockIdx.x, tid = threadIdx.x;
    const float* sp = scores + (size_t)b * N;
    const float* bp = boxes + (size_t)b * N * 4;

    __shared__ unsigned hist[256];
    __shared__ unsigned s_prefix, s_want, s_cnt;
    __shared__ unsigned long long cand[2048];
    __shared__ float bx0[1024], by0[1024], bx1[1024], by1[1024], sar[1024], ssc[1024];
    __shared__ int keep[1024];

    if (tid == 0) { s_prefix = 0u; s_want = (unsigned)k; }
    // 4-pass MSB radix select for the k-th largest key
    for (int pass = 0; pass < 4; ++pass) {
        int shift = 24 - pass * 8;
        if (tid < 256) hist[tid] = 0u;
        __syncthreads();
        unsigned pref = s_prefix;
        for (int i = tid; i < N; i += 1024) {
            unsigned u = fkey(sp[i]);
            bool match = (pass == 0) || ((u >> (shift + 8)) == (pref >> (shift + 8)));
            if (match) atomicAdd(&hist[(u >> shift) & 255], 1u);
        }
        __syncthreads();
        if (tid == 0) {
            unsigned want = s_want, cum = 0; int bin = 0;
            for (int bb = 255; bb >= 0; --bb) {
                if (cum + hist[bb] >= want) { bin = bb; break; }
                cum += hist[bb];
            }
            s_prefix = pref | ((unsigned)bin << shift);
            s_want = want - cum;
        }
        __syncthreads();
    }
    unsigned T = s_prefix;
    if (tid == 0) s_cnt = 0u;
    __syncthreads();
    for (int i = tid; i < N; i += 1024) {
        unsigned u = fkey(sp[i]);
        if (u >= T) {
            unsigned pos = atomicAdd(&s_cnt, 1u);
            if (pos < 2048)
                cand[pos] = (((unsigned long long)(~u)) << 32) | (unsigned)i; // asc sort => score desc, idx asc
        }
    }
    __syncthreads();
    unsigned nc = s_cnt; if (nc > 2048) nc = 2048;
    for (int i = tid; i < 2048; i += 1024)
        if (i >= (int)nc) cand[i] = ~0ULL;     // sentinel -> end
    __syncthreads();
    // bitonic sort ascending (2048 elems, 1024 threads)
    for (int kk = 2; kk <= 2048; kk <<= 1)
        for (int j = kk >> 1; j > 0; j >>= 1) {
            for (int i = tid; i < 2048; i += 1024) {
                int ixj = i ^ j;
                if (ixj > i) {
                    bool up = ((i & kk) == 0);
                    unsigned long long a = cand[i], c = cand[ixj];
                    bool sw = up ? (a > c) : (a < c);
                    if (sw) { cand[i] = c; cand[ixj] = a; }
                }
            }
            __syncthreads();
        }
    // gather top-k into LDS
    for (int s = tid; s < k; s += 1024) {
        unsigned idx = (unsigned)(cand[s] & 0xFFFFFFFFu);
        float sc = sp[idx];
        const float* bo = bp + (size_t)idx * 4;
        float X1 = bo[0], Y1 = bo[1], X2 = bo[2], Y2 = bo[3];
        bx0[s] = X1; by0[s] = Y1; bx1[s] = X2; by1[s] = Y2;
        sar[s] = (X2 - X1) * (Y2 - Y1);
        ssc[s] = sc; keep[s] = 1;
    }
    __syncthreads();
    // greedy NMS: sequential i, parallel j (j > i; forward-only suppression is equivalent)
    for (int i = 0; i < k; ++i) {
        if (keep[i]) {
            float xi1 = bx0[i], yi1 = by0[i], xi2 = bx1[i], yi2 = by1[i], ai = sar[i];
            int j = i + 1 + tid;
            if (j < k) {
                float lx = fmaxf(xi1, bx0[j]);
                float ly = fmaxf(yi1, by0[j]);
                float rx = fminf(xi2, bx1[j]);
                float ry = fminf(yi2, by1[j]);
                float iw = fmaxf(rx - lx, 0.0f);
                float ih = fmaxf(ry - ly, 0.0f);
                float inter = iw * ih;
                float iou = inter / (ai + sar[j] - inter + 1e-9f);
                if (iou > NMS_THR) keep[j] = 0;
            }
        }
        __syncthreads();
    }
    for (int s = tid; s < k; s += 1024) {
        int o = b * TOT + off + s;
        stg_s[o] = keep[s] ? ssc[s] : NEG_INF_F;
        stg_b[(size_t)o * 4 + 0] = bx0[s];
        stg_b[(size_t)o * 4 + 1] = by0[s];
        stg_b[(size_t)o * 4 + 2] = bx1[s];
        stg_b[(size_t)o * 4 + 3] = by1[s];
    }
}

// ---------------- final top-1000 over concatenated levels ----------------
__global__ void __launch_bounds__(1024)
final_topk(const float* __restrict__ stg_s, const float* __restrict__ stg_b,
           float* __restrict__ out, int TOT) {
    int b = blockIdx.x, tid = threadIdx.x;
    __shared__ unsigned skey[8192];
    __shared__ unsigned short sval[8192];
    for (int i = tid; i < 8192; i += 1024) {
        if (i < TOT) {
            skey[i] = ~fkey(stg_s[b * TOT + i]);   // ascending sort => score descending
            sval[i] = (unsigned short)i;
        } else { skey[i] = 0xFFFFFFFFu; sval[i] = 0; }
    }
    __syncthreads();
    for (int kk = 2; kk <= 8192; kk <<= 1)
        for (int j = kk >> 1; j > 0; j >>= 1) {
            for (int i = tid; i < 8192; i += 1024) {
                int ixj = i ^ j;
                if (ixj > i) {
                    bool up = ((i & kk) == 0);
                    unsigned a = skey[i], c = skey[ixj];
                    bool sw = up ? (a > c) : (a < c);
                    if (sw) {
                        skey[i] = c; skey[ixj] = a;
                        unsigned short tv = sval[i]; sval[i] = sval[ixj]; sval[ixj] = tv;
                    }
                }
            }
            __syncthreads();
        }
    for (int s = tid; s < 1000; s += 1024) {
        int idx = sval[s];
        int src = b * TOT + idx;
        float* o = out + (size_t)(b * 1000 + s) * 5;
        o[0] = stg_b[(size_t)src * 4 + 0];
        o[1] = stg_b[(size_t)src * 4 + 1];
        o[2] = stg_b[(size_t)src * 4 + 2];
        o[3] = stg_b[(size_t)src * 4 + 3];
        o[4] = stg_s[src];
    }
}

// ---------------- host ----------------
extern "C" void kernel_launch(void* const* d_in, const int* in_sizes, int n_in,
                              void* d_out, int out_size, void* d_ws, size_t ws_size,
                              hipStream_t stream) {
    (void)in_sizes; (void)n_in; (void)out_size; (void)ws_size;
    const float* feats[5] = {(const float*)d_in[0], (const float*)d_in[1], (const float*)d_in[2],
                             (const float*)d_in[3], (const float*)d_in[4]};
    const float* conv_w = (const float*)d_in[5];
    const float* conv_b = (const float*)d_in[6];
    const float* cls_w  = (const float*)d_in[7];
    const float* cls_b  = (const float*)d_in[8];
    const float* bbox_w = (const float*)d_in[9];
    const float* bbox_b = (const float*)d_in[10];
    float* out = (float*)d_out;

    char* ws = (char*)d_ws;
    size_t off = 0;
    auto take = [&](size_t bytes) -> char* {
        char* p = ws + off;
        off += (bytes + 255) & ~(size_t)255;
        return p;
    };
    _Float16* xf16 = (_Float16*)take(2ull * 258 * 258 * 256 * 2);   // padded NHWC f16 (max level)
    _Float16* tf16 = (_Float16*)take(2ull * 256 * 256 * 256 * 2);   // NHWC f16 conv output
    _Float16* wB   = (_Float16*)take(256ull * 2304 * 2);            // conv weights B^T
    _Float16* hB   = (_Float16*)take(16ull * 256 * 2);              // head weights
    float* hbias   = (float*)take(16 * 4);
    float* scores  = (float*)take(2ull * 196608 * 4);
    float* boxes   = (float*)take(2ull * 196608 * 4 * 4);
    const int TOT = 4768;                                           // 1000*4 + 768
    float* stg_s = (float*)take(2ull * TOT * 4);
    float* stg_b = (float*)take(2ull * TOT * 4 * 4);

    prep_conv_weights<<<(256 * 2304 + 255) / 256, 256, 0, stream>>>(conv_w, wB);
    prep_head_weights<<<16, 256, 0, stream>>>(cls_w, cls_b, bbox_w, bbox_b, hB, hbias);

    const int   Hs[5]      = {256, 128, 64, 32, 16};
    const float strides[5] = {4.f, 8.f, 16.f, 32.f, 64.f};
    const float asizes[5]  = {32.f, 64.f, 128.f, 256.f, 512.f};
    int lvl_off = 0;
    for (int l = 0; l < 5; ++l) {
        int H = Hs[l], W = Hs[l], HW = H * W;
        int Hp = H + 2, Wp = W + 2;
        int halopix = 2 * Hp * Wp;
        pad_halo_zero<<<(halopix + 255) / 256, 256, 0, stream>>>(xf16, H, W);
        int total = 2 * 256 * HW;
        nchw_to_nhwc_pad_f16<<<(total + 255) / 256, 256, 0, stream>>>(feats[l], xf16, H, W);
        int P = 2 * HW;
        // conv: waves = (P/32) m-tiles * 4 n-groups; 8 waves/block
        int waves = (P / 32) * 4;
        conv3x3_wmma<<<waves / 8, 256, 0, stream>>>(xf16, wB, conv_b, tf16, H, W);
        int m_tiles = P / 16;
        head_wmma<<<(m_tiles + 7) / 8, 256, 0, stream>>>(tf16, hB, hbias, scores, boxes,
                                                         H, W, strides[l], asizes[l]);
        int N = HW * 3;
        int k = N < 1000 ? N : 1000;
        topk_nms<<<2, 1024, 0, stream>>>(scores, boxes, N, k, stg_s, stg_b, lvl_off, TOT);
        lvl_off += k;
    }
    final_topk<<<2, 1024, 0, stream>>>(stg_s, stg_b, out, TOT);
}
